// Base_52321291600377
// MI455X (gfx1250) — matrix operand order repaired and checked by reading, not compiled, not run on hardware
//
#include <hip/hip_runtime.h>
#include <hip/hip_bf16.h>
#include <math.h>

// ---------------------------------------------------------------------------
// Problem constants (from reference): B=4, T=1024, D=1024, H=16, Dh=64, L=12
// ---------------------------------------------------------------------------
static constexpr int kB  = 4;
static constexpr int kT  = 1024;
static constexpr int kD  = 1024;
static constexpr int kH  = 16;
static constexpr int kDh = 64;
static constexpr int kL  = 12;
static constexpr int kBT = kB * kT;          // 4096 rows
static constexpr float kEps = 1e-5f;

typedef __bf16 bf16;
typedef __attribute__((ext_vector_type(16))) __bf16    v16bf;
typedef __attribute__((ext_vector_type(8)))  float     v8f;
typedef __attribute__((ext_vector_type(4)))  int       v4i;
typedef __attribute__((ext_vector_type(8)))  int       v8i;
typedef __attribute__((ext_vector_type(4)))  unsigned  v4u;

// pointer flavors the async builtin wants (param0 printed as 'int4 __device__ *')
typedef __attribute__((address_space(1))) v4i* as1_v4i;
typedef __attribute__((address_space(3))) v4i* as3_v4i;

// ---------------------------------------------------------------------------
// CDNA5 async global->LDS copy (ASYNCcnt-tracked, bypasses VGPRs) + waits
// ---------------------------------------------------------------------------
__device__ __forceinline__ void async_copy_b128(const bf16* gsrc, bf16* ldst) {
#if __has_builtin(__builtin_amdgcn_global_load_async_to_lds_b128)
    __builtin_amdgcn_global_load_async_to_lds_b128(
        (as1_v4i)(uintptr_t)gsrc,
        (as3_v4i)(uintptr_t)(unsigned)(uintptr_t)ldst,
        0, 0);
#else
    unsigned loff = (unsigned)(uintptr_t)ldst;
    asm volatile("global_load_async_to_lds_b128 %0, %1, off"
                 :: "v"(loff), "v"((unsigned long long)(uintptr_t)gsrc)
                 : "memory");
#endif
}

template<int N>
__device__ __forceinline__ void wait_asynccnt() {
#if __has_builtin(__builtin_amdgcn_s_wait_asynccnt)
    __builtin_amdgcn_s_wait_asynccnt(N);
#else
    asm volatile("s_wait_asynccnt %0" :: "n"(N) : "memory");
#endif
}

// ---------------------------------------------------------------------------
// CDNA5 Tensor Data Mover: one descriptor DMAs a whole 2D tile global->LDS.
// D# packing per CDNA5 ISA:
//   g0: count=1 | lds_addr | global_addr[56:0] | type=2
//   g1: data_size=1(2B), pad_enable, pad_interval=4 (32 DW = one 64-half row),
//       pad_amount=3 (4 DW = 8 halves)  -> LDS row stride 72 halves (= LSTR)
//       tensor_dim0/1, tile_dim0/1, tensor_dim0_stride
//   g2/g3 (+ trailing group on this toolchain): zero (2D tensor)
// This toolchain's builtin is the 6-arg flavor:
//   (uint32x4 g0, int32x8 g1, int32x4, int32x4, int32x8, i32 cpol)
// ---------------------------------------------------------------------------
#if __has_builtin(__builtin_amdgcn_tensor_load_to_lds)
#define TDM_OK 1
#else
#define TDM_OK 0
#endif

template<int TILE0, int TILE1>
__device__ __forceinline__ void tdm_load_2d(const bf16* gsrc, unsigned lds_off,
                                            int dim0, int dim1, int ld) {
#if TDM_OK
    const unsigned long long ga = (unsigned long long)(uintptr_t)gsrc;
    v4u g0;
    g0[0] = 1u;                                            // count=1, user D#
    g0[1] = lds_off;                                       // LDS byte address
    g0[2] = (unsigned)ga;                                  // global_addr[31:0]
    g0[3] = (unsigned)((ga >> 32) & 0x1FFFFFFu) | (2u << 30); // addr[56:32] | type=2
    v8i g1;
    g1[0] = (int)((1u << 16)        // data_size = 1 (2 bytes)
                | (1u << 20)        // pad_enable
                | (4u << 22)        // pad_interval: 32 DWORDs
                | (3u << 25));      // pad_amount:   4 DWORDs
    g1[1] = (int)(((unsigned)dim0 & 0xFFFFu) << 16);               // tensor_dim0[15:0]
    g1[2] = (int)(((unsigned)dim0 >> 16) | (((unsigned)dim1 & 0xFFFFu) << 16));
    g1[3] = (int)(((unsigned)dim1 >> 16) | ((unsigned)TILE0 << 16)); // tile_dim0
    g1[4] = (int)(unsigned)TILE1;                                   // tile_dim1 (tile_dim2=0)
    g1[5] = ld;                                                     // dim0_stride[31:0]
    g1[6] = 0;
    g1[7] = 0;
    v4i gz4 = {0, 0, 0, 0};
    v8i gz8 = {0, 0, 0, 0, 0, 0, 0, 0};
    __builtin_amdgcn_tensor_load_to_lds(g0, g1, gz4, gz4, gz8, 0);
#else
    (void)gsrc; (void)lds_off; (void)dim0; (void)dim1; (void)ld;
#endif
}

template<int N>
__device__ __forceinline__ void wait_tensorcnt() {
#if __has_builtin(__builtin_amdgcn_s_wait_tensorcnt)
    __builtin_amdgcn_s_wait_tensorcnt((short)N);
#else
    asm volatile("s_wait_tensorcnt %0" :: "n"(N) : "memory");
#endif
}

// ---------------------------------------------------------------------------
// fp32 W[K,N] -> bf16 WT[N,K] tiled transpose-convert (weights then sit in L2
// K-contiguous so GEMM B tiles stage as single TDM descriptors)
// ---------------------------------------------------------------------------
__global__ __launch_bounds__(256) void cvt_transpose(const float* __restrict__ in,
                                                     bf16* __restrict__ out,
                                                     int Kd, int Nd) {
    __shared__ bf16 tile[32][33];
    const int tx = threadIdx.x & 31;
    const int ty = threadIdx.x >> 5;           // 8 rows per pass
    const int n0 = blockIdx.x * 32;
    const int k0 = blockIdx.y * 32;
#pragma unroll
    for (int r = 0; r < 32; r += 8)
        tile[ty + r][tx] = (bf16)in[(size_t)(k0 + ty + r) * Nd + n0 + tx];   // tile[k][n]
    __syncthreads();
#pragma unroll
    for (int r = 0; r < 32; r += 8)
        out[(size_t)(n0 + ty + r) * Kd + k0 + tx] = tile[tx][ty + r];        // WT[n][k]
}

// ---------------------------------------------------------------------------
// LayerNorm over D=1024, times row mask, output bf16.  One row per 256-thread block.
// ---------------------------------------------------------------------------
__global__ __launch_bounds__(256) void ln_kernel(const float* __restrict__ x,
                                                 const float* __restrict__ mrow,
                                                 const float* __restrict__ g,
                                                 const float* __restrict__ b,
                                                 bf16* __restrict__ out) {
    const int row = blockIdx.x;            // 0..B*T-1
    const int tid = threadIdx.x;
    const float* xr = x + (size_t)row * kD;
    __shared__ float rs[256], rq[256];
    float v[4], s = 0.f, q = 0.f;
#pragma unroll
    for (int i = 0; i < 4; i++) { v[i] = xr[tid + i * 256]; s += v[i]; q += v[i] * v[i]; }
    rs[tid] = s; rq[tid] = q; __syncthreads();
    for (int off = 128; off > 0; off >>= 1) {
        if (tid < off) { rs[tid] += rs[tid + off]; rq[tid] += rq[tid + off]; }
        __syncthreads();
    }
    const float mu  = rs[0] * (1.0f / kD);
    const float var = rq[0] * (1.0f / kD) - mu * mu;
    const float inv = rsqrtf(var + kEps);
    const float mr  = mrow[row];
#pragma unroll
    for (int i = 0; i < 4; i++) {
        int c = tid + i * 256;
        out[(size_t)row * kD + c] = (bf16)(((v[i] - mu) * inv * g[c] + b[c]) * mr);
    }
}

// ---------------------------------------------------------------------------
// Causal softmax over keys, f32 scores in, bf16 probs out (zeros above diagonal).
// ---------------------------------------------------------------------------
__global__ __launch_bounds__(256) void softmax_kernel(const float* __restrict__ s,
                                                      bf16* __restrict__ p) {
    const int row = blockIdx.x;            // 0..B*H*T-1
    const int i   = row & (kT - 1);        // query index (causal limit)
    const int tid = threadIdx.x;
    const float* sr = s + (size_t)row * kT;
    bf16* pr = p + (size_t)row * kT;
    __shared__ float red[256];

    float mx = -3.402823466e38f;
    for (int j = tid; j < kT; j += 256) if (j <= i) mx = fmaxf(mx, sr[j]);
    red[tid] = mx; __syncthreads();
    for (int off = 128; off > 0; off >>= 1) {
        if (tid < off) red[tid] = fmaxf(red[tid], red[tid + off]);
        __syncthreads();
    }
    mx = red[0]; __syncthreads();

    float sum = 0.f;
    for (int j = tid; j < kT; j += 256) if (j <= i) sum += expf(sr[j] - mx);
    red[tid] = sum; __syncthreads();
    for (int off = 128; off > 0; off >>= 1) {
        if (tid < off) red[tid] += red[tid + off];
        __syncthreads();
    }
    const float inv = 1.0f / red[0];

    for (int j = tid; j < kT; j += 256)
        pr[j] = (bf16)((j <= i) ? expf(sr[j] - mx) * inv : 0.0f);
}

// ---------------------------------------------------------------------------
// Tiled bf16 WMMA GEMM, f32 accumulate, TDM/async double-buffered LDS staging.
//   MODE 0: plain C = A @ B                      (blockIdx.z unused)
//   MODE 1: attention scores S = Q @ K^T, batched over (b,h), causal tile skip
//   MODE 2: attention PV     O = P @ V,   batched over (b,h)
//   BTRANS: B stored N-major (K contiguous) -> TDM tile staging;
//           else B is K x N and transposed through VGPRs into LDS (PV only).
// Fragment layouts per CDNA5 ISA 7.12.2 (wave32):
//   A 16x32 bf16: lane row = lane%16, K = {kb..kb+7, kb+16..kb+23}, kb=8*(lane/16)
//   B 32x16 bf16: lane col = lane%16, K = {kb..kb+7, kb+16..kb+23}, kb=8*(lane/16)
//   C 16x16 f32 : lane col = lane%16, rows = 8*(lane/16) + r, r=0..7
// ---------------------------------------------------------------------------
template<int MODE, int BM, int BN, int WM, int WN, bool BTRANS,
         bool BIAS, bool RESID, bool GELU_, bool CAUSAL, bool MASKR, bool OUTBF16>
__global__ __launch_bounds__((BM / WM) * (BN / WN) * 32)
void gemm_wmma(const bf16* __restrict__ A, const bf16* __restrict__ Bm,
               const float* __restrict__ bias, const float* __restrict__ resid,
               const float* __restrict__ rowmask,
               float* __restrict__ Cf, bf16* __restrict__ Cb,
               int M, int N, int K, int lda, int ldb, int ldc, float scale) {
    constexpr int BK   = 64;                       // two 16x16x32 K-substeps
    constexpr int NT   = (BM / WM) * (BN / WN) * 32;
    constexpr int FM   = WM / 16, FN = WN / 16;
    constexpr int LSTR = BK + 8;                   // padded LDS stride (halves)
    constexpr int NTDM = BTRANS ? 2 : 1;           // TDM descriptors per staged tile
    // per-wave in-flight async-copy count (fallback path)
    constexpr int NASY = (BM * BK / 8) / NT + (BTRANS ? (BN * BK / 8) / NT : 0);

    __shared__ __align__(16) bf16 As[2][BM][LSTR];
    __shared__ __align__(16) bf16 Bs[2][BN][LSTR];

    const int tid  = threadIdx.x;
    const int lane = tid & 31;
    const int wid  = tid >> 5;
    constexpr int WAVES_N = BN / WN;
    const int wm = wid / WAVES_N;
    const int wn = wid % WAVES_N;

    const int n0 = blockIdx.x * BN;
    const int m0 = blockIdx.y * BM;

    if (MODE == 1) {                               // scores: Q @ K^T, z = b*H+h
        const int bh = blockIdx.z, b = bh >> 4, h = bh & 15;
        const size_t off = (size_t)b * kT * (3 * kD) + (size_t)h * kDh;
        A  += off;                                 // Q rows
        Bm += off;                                 // K rows (k-contiguous)
        Cf += (size_t)bh * kT * kT;
        if (n0 > m0 + BM - 1) return;              // tile fully above diagonal
    } else if (MODE == 2) {                        // PV: P @ V
        const int bh = blockIdx.z, b = bh >> 4, h = bh & 15;
        A  += (size_t)bh * kT * kT;                // probs
        Bm += (size_t)b * kT * (3 * kD) + (size_t)h * kDh;   // V rows
        Cb += (size_t)b * kT * kD + (size_t)h * kDh;         // O at head offset
    }

    // ---- tile staging ----
    auto stage = [&](int k0, int buf) {
#if TDM_OK
        if (wid == 0) {                            // one wave drives the DMA engine
            tdm_load_2d<BK, BM>(A + (size_t)m0 * lda + k0,
                                (unsigned)(uintptr_t)&As[buf][0][0], lda, M, lda);
            if (BTRANS)
                tdm_load_2d<BK, BN>(Bm + (size_t)n0 * ldb + k0,
                                    (unsigned)(uintptr_t)&Bs[buf][0][0], ldb, N, ldb);
        }
#else
        constexpr int A_VECS = BM * BK / 8;
        for (int e = tid; e < A_VECS; e += NT) {
            const int row = e >> 3;                // BK/8 == 8
            const int c   = (e & 7) * 8;
            async_copy_b128(A + (size_t)(m0 + row) * lda + k0 + c, &As[buf][row][c]);
        }
        if (BTRANS) {
            constexpr int B_VECS = BN * BK / 8;
            for (int e = tid; e < B_VECS; e += NT) {
                const int row = e >> 3;
                const int c   = (e & 7) * 8;
                async_copy_b128(Bm + (size_t)(n0 + row) * ldb + k0 + c, &Bs[buf][row][c]);
            }
        }
#endif
        if (!BTRANS) {                             // transpose K x N -> Bs[n][k]
            constexpr int B_PAIRS = BN * BK / 2;
            for (int e = tid; e < B_PAIRS; e += NT) {
                const int kk = e / (BN / 2);
                const int n  = (e % (BN / 2)) * 2;
                union { unsigned u; bf16 h[2]; } cv;
                cv.u = *(const unsigned*)(Bm + (size_t)(k0 + kk) * ldb + n0 + n);
                Bs[buf][n][kk]     = cv.h[0];
                Bs[buf][n + 1][kk] = cv.h[1];
            }
        }
    };

    auto wait_stage = [&](bool hasNext) {
#if TDM_OK
        if (wid == 0) {                            // in-order: <=NTDM left => cur done
            if (hasNext) wait_tensorcnt<NTDM>();
            else         wait_tensorcnt<0>();
        }
#else
        if (hasNext) wait_asynccnt<NASY>();
        else         wait_asynccnt<0>();
#endif
    };

    v8f acc[FM][FN];
#pragma unroll
    for (int i = 0; i < FM; i++)
#pragma unroll
        for (int j = 0; j < FN; j++) acc[i][j] = {};

    union FragBf { uint4 q[2]; v16bf v; };
    const int r16 = lane & 15;
    const int hi  = lane >> 4;

    stage(0, 0);                                   // prologue: tile 0 in flight
    int buf = 0;
    for (int k0 = 0; k0 < K; k0 += BK) {
        const bool hasNext = (k0 + BK < K);
        if (hasNext) stage(k0 + BK, buf ^ 1);      // overlap next tile with compute
        wait_stage(hasNext);                       // current tile resident
        __syncthreads();

#pragma unroll
        for (int ks = 0; ks < BK; ks += 32) {
            FragBf af[FM], bfrag[FN];
            const int akb = ks + hi * 8;
#pragma unroll
            for (int i = 0; i < FM; i++) {
                const int row = wm * WM + i * 16 + r16;
                af[i].q[0] = *(const uint4*)&As[buf][row][akb];
                af[i].q[1] = *(const uint4*)&As[buf][row][akb + 16];
            }
            const int bkb = ks + hi * 8;
#pragma unroll
            for (int j = 0; j < FN; j++) {
                const int col = wn * WN + j * 16 + r16;
                bfrag[j].q[0] = *(const uint4*)&Bs[buf][col][bkb];
                bfrag[j].q[1] = *(const uint4*)&Bs[buf][col][bkb + 16];
            }
#pragma unroll
            for (int i = 0; i < FM; i++)
#pragma unroll
                for (int j = 0; j < FN; j++)
                    acc[i][j] = __builtin_amdgcn_wmma_f32_16x16x32_bf16(
                        false, af[i].v, false, bfrag[j].v,
                        (short)0, acc[i][j], false, false);
        }
        __syncthreads();                           // reads done before next overwrite
        buf ^= 1;
    }

    // ---- fused epilogue ----
    const int hi8 = hi * 8;
#pragma unroll
    for (int i = 0; i < FM; i++) {
#pragma unroll
        for (int j = 0; j < FN; j++) {
            const int gn  = n0 + wn * WN + j * 16 + r16;
            const int gm0 = m0 + wm * WM + i * 16 + hi8;
            const float bv = BIAS ? bias[gn] : 0.0f;
#pragma unroll
            for (int r = 0; r < 8; r++) {
                const int gm = gm0 + r;
                float v = acc[i][j][r] * scale + bv;
                if (GELU_)  v = 0.5f * v * (1.0f + erff(v * 0.70710678118654752f));
                if (CAUSAL) { if (gn > gm) v = -3.402823466e38f; }
                if (RESID)  v += resid[(size_t)gm * ldc + gn];
                if (MASKR)  v *= rowmask[gm];
                if (OUTBF16) Cb[(size_t)gm * ldc + gn] = (bf16)v;
                else         Cf[(size_t)gm * ldc + gn] = v;
            }
        }
    }
}

// ---------------------------------------------------------------------------
// Host-side orchestration
// ---------------------------------------------------------------------------
extern "C" void kernel_launch(void* const* d_in, const int* in_sizes, int n_in,
                              void* d_out, int out_size, void* d_ws, size_t ws_size,
                              hipStream_t stream) {
    const float* x_in  = (const float*)d_in[0];
    const float* mvec  = (const float*)d_in[1];   // (B,T,1) row mask
    const float* ln1_g = (const float*)d_in[2];
    const float* ln1_b = (const float*)d_in[3];
    const float* wqkv  = (const float*)d_in[4];
    const float* wout  = (const float*)d_in[5];
    const float* bout  = (const float*)d_in[6];
    const float* ln2_g = (const float*)d_in[7];
    const float* ln2_b = (const float*)d_in[8];
    const float* w1    = (const float*)d_in[9];
    const float* b1    = (const float*)d_in[10];
    const float* w2    = (const float*)d_in[11];
    const float* b2    = (const float*)d_in[12];
    float* x = (float*)d_out;                     // running residual stream

    const size_t DSQ = (size_t)kD * kD;

    // workspace carve-up (per-layer buffers reused across layers)
    char*  ws  = (char*)d_ws;
    size_t cur = 0;
    auto alloc = [&](size_t bytes) -> void* {
        void* p = ws + cur;
        cur += (bytes + 255) & ~(size_t)255;
        return p;
    };
    bf16* wbqkv  = (bf16*)alloc(3 * DSQ * sizeof(bf16));            // 6 MB  (N-major)
    bf16* wbout  = (bf16*)alloc(DSQ * sizeof(bf16));                // 2 MB  (N-major)
    bf16* wb1    = (bf16*)alloc(4 * DSQ * sizeof(bf16));            // 8 MB  (N-major)
    bf16* wb2    = (bf16*)alloc(4 * DSQ * sizeof(bf16));            // 8 MB  (N-major)
    bf16* hb     = (bf16*)alloc((size_t)kBT * kD * sizeof(bf16));   // 8 MB
    bf16* qkvb   = (bf16*)alloc((size_t)kBT * 3 * kD * sizeof(bf16)); // 24 MB
    float* scoresf = (float*)alloc((size_t)kB * kH * kT * kT * sizeof(float)); // 256 MB
    bf16* probsb = (bf16*)alloc((size_t)kB * kH * kT * kT * sizeof(bf16));     // 128 MB
    bf16* attnob = (bf16*)alloc((size_t)kBT * kD * sizeof(bf16));   // 8 MB
    bf16* h2b    = (bf16*)alloc((size_t)kBT * kD * sizeof(bf16));   // 8 MB
    bf16* ffb    = (bf16*)alloc((size_t)kBT * 4 * kD * sizeof(bf16)); // 32 MB
    (void)ws_size; (void)n_in; (void)out_size; (void)in_sizes;

    // x <- input (residual stream lives in d_out)
    (void)hipMemcpyAsync(x, x_in, (size_t)kBT * kD * sizeof(float),
                         hipMemcpyDeviceToDevice, stream);

    const dim3 blk(256);
    const float iscale = 0.125f;                 // 1/sqrt(Dh)

    for (int l = 0; l < kL; l++) {
        const float* Wqkv = wqkv + (size_t)l * 3 * DSQ;
        const float* Wout = wout + (size_t)l * DSQ;
        const float* W1   = w1   + (size_t)l * 4 * DSQ;
        const float* W2   = w2   + (size_t)l * 4 * DSQ;

        // --- convert+transpose this layer's weights fp32 -> bf16 N-major
        //     (24 MB/layer: stays hot in the 192 MB global L2 across re-reads) ---
        cvt_transpose<<<dim3(3 * kD / 32, kD / 32), blk, 0, stream>>>(Wqkv, wbqkv, kD, 3 * kD);
        cvt_transpose<<<dim3(kD / 32, kD / 32),     blk, 0, stream>>>(Wout, wbout, kD, kD);
        cvt_transpose<<<dim3(4 * kD / 32, kD / 32), blk, 0, stream>>>(W1, wb1, kD, 4 * kD);
        cvt_transpose<<<dim3(kD / 32, 4 * kD / 32), blk, 0, stream>>>(W2, wb2, 4 * kD, kD);

        // --- h = LN1(x) * m  (bf16) ---
        ln_kernel<<<dim3(kBT), blk, 0, stream>>>(x, mvec, ln1_g + (size_t)l * kD,
                                                 ln1_b + (size_t)l * kD, hb);

        // --- qkv = h @ Wqkv  (4096 x 3072 x 1024) -> bf16 ---
        gemm_wmma<0, 128, 128, 32, 64, true, false, false, false, false, false, true>
            <<<dim3(3 * kD / 128, kBT / 128, 1), blk, 0, stream>>>(
                hb, wbqkv, nullptr, nullptr, nullptr, nullptr, qkvb,
                kBT, 3 * kD, kD, kD, kD, 3 * kD, 1.0f);

        // --- scores = (Q @ K^T) * scale, causal  (64 x [1024 x 1024 x 64]) ---
        gemm_wmma<1, 128, 128, 32, 64, true, false, false, false, true, false, false>
            <<<dim3(kT / 128, kT / 128, kB * kH), blk, 0, stream>>>(
                qkvb, qkvb + kD, nullptr, nullptr, nullptr, scoresf, nullptr,
                kT, kT, kDh, 3 * kD, 3 * kD, kT, iscale);

        // --- P = causal softmax(scores) -> bf16 ---
        softmax_kernel<<<dim3(kB * kH * kT), blk, 0, stream>>>(scoresf, probsb);

        // --- O = P @ V  (64 x [1024 x 64 x 1024]) -> bf16 at head offset ---
        gemm_wmma<2, 128, 64, 32, 32, false, false, false, false, false, false, true>
            <<<dim3(kDh / 64, kT / 128, kB * kH), blk, 0, stream>>>(
                probsb, qkvb + 2 * kD, nullptr, nullptr, nullptr, nullptr, attnob,
                kT, kDh, kT, kT, 3 * kD, kD, 1.0f);

        // --- x = (x + (O @ Wout + bout)) * m  (4096 x 1024 x 1024) ---
        gemm_wmma<0, 128, 128, 32, 64, true, true, true, false, false, true, false>
            <<<dim3(kD / 128, kBT / 128, 1), blk, 0, stream>>>(
                attnob, wbout, bout + (size_t)l * kD, x, mvec, x, nullptr,
                kBT, kD, kD, kD, kD, kD, 1.0f);

        // --- h2 = LN2(x) * m ---
        ln_kernel<<<dim3(kBT), blk, 0, stream>>>(x, mvec, ln2_g + (size_t)l * kD,
                                                 ln2_b + (size_t)l * kD, h2b);

        // --- ff = gelu(h2 @ W1 + b1)  (4096 x 4096 x 1024) -> bf16 ---
        gemm_wmma<0, 128, 128, 32, 64, true, true, false, true, false, false, true>
            <<<dim3(4 * kD / 128, kBT / 128, 1), blk, 0, stream>>>(
                h2b, wb1, b1 + (size_t)l * 4 * kD, nullptr, nullptr, nullptr, ffb,
                kBT, 4 * kD, kD, kD, kD, 4 * kD, 1.0f);

        // --- x = (x + (ff @ W2 + b2)) * m  (4096 x 1024 x 4096) ---
        gemm_wmma<0, 128, 128, 32, 64, true, true, true, false, false, true, false>
            <<<dim3(kD / 128, kBT / 128, 1), blk, 0, stream>>>(
                ffb, wb2, b2 + (size_t)l * kD, x, mvec, x, nullptr,
                kBT, kD, 4 * kD, 4 * kD, 4 * kD, kD, 1.0f);
    }
}
